// SimpleCrossAttention_13185549598923
// MI455X (gfx1250) — compile-verified, matrix operand order checked by reading, not run on hardware
//
#include <hip/hip_runtime.h>

#define DIM     1024
#define NQ      1024
#define NC      4096
#define NHEADS  16
#define HD      64

typedef _Float16 v16h __attribute__((ext_vector_type(16)));
typedef _Float16 v8h  __attribute__((ext_vector_type(8)));
typedef _Float16 v4h  __attribute__((ext_vector_type(4)));
typedef float    v8f  __attribute__((ext_vector_type(8)));
typedef float    v4f  __attribute__((ext_vector_type(4)));
typedef unsigned int v4u __attribute__((ext_vector_type(4)));
typedef int      v8i  __attribute__((ext_vector_type(8)));
typedef int      v4i  __attribute__((ext_vector_type(4)));

// Build a 16-half WMMA fragment from two contiguous 8-half (16B) chunks.
// Per CDNA5 16-bit A layout: lane L holds row (L&15); chunk base (L>>4)*8,
// elements {cb..cb+7} and {cb+16..cb+23} of the 32-wide K slice.
__device__ __forceinline__ v16h frag_ld(const _Float16* p0, const _Float16* p1) {
    v8h lo = *(const v8h*)p0;
    v8h hi = *(const v8h*)p1;
    v16h r;
#pragma unroll
    for (int i = 0; i < 8; ++i) { r[i] = lo[i]; r[i + 8] = hi[i]; }
    return r;
}

// ---------------------------------------------------------------------------
// Fused LayerNorm + f16 cast. One block (256 threads) per row of 1024 floats.
// ---------------------------------------------------------------------------
__global__ __launch_bounds__(256)
void ln_kernel(const float* __restrict__ x,
               const float* __restrict__ g,
               const float* __restrict__ b,
               float* __restrict__ y32,          // nullable (residual copy)
               _Float16* __restrict__ y16)
{
    const int row  = blockIdx.x;
    const int tid  = threadIdx.x;
    const size_t base = (size_t)row * DIM;

    v4f xv = *(const v4f*)(x + base + tid * 4);
    float s = xv[0] + xv[1] + xv[2] + xv[3];
    float q = xv[0]*xv[0] + xv[1]*xv[1] + xv[2]*xv[2] + xv[3]*xv[3];
#pragma unroll
    for (int off = 16; off >= 1; off >>= 1) {
        s += __shfl_xor(s, off, 32);
        q += __shfl_xor(q, off, 32);
    }
    __shared__ float ps[8], pq[8];
    const int wid = tid >> 5, lane = tid & 31;
    if (lane == 0) { ps[wid] = s; pq[wid] = q; }
    __syncthreads();
    float S = 0.f, Q = 0.f;
#pragma unroll
    for (int i = 0; i < 8; ++i) { S += ps[i]; Q += pq[i]; }
    const float mu   = S * (1.0f / DIM);
    const float var  = Q * (1.0f / DIM) - mu * mu;
    const float rstd = rsqrtf(var + 1e-5f);

    v4f gg = *(const v4f*)(g + tid * 4);
    v4f bb = *(const v4f*)(b + tid * 4);
    v4f yv; v4h yh;
#pragma unroll
    for (int i = 0; i < 4; ++i) {
        float y = (xv[i] - mu) * rstd * gg[i] + bb[i];
        yv[i] = y; yh[i] = (_Float16)y;
    }
    if (y32) *(v4f*)(y32 + base + tid * 4) = yv;
    *(v4h*)(y16 + base + tid * 4) = yh;
}

// ---------------------------------------------------------------------------
// WMMA GEMM: Y[M,1024] = X_f16[M,1024] @ W_f32[1024,1024]^T + bias (+resid)
// Block tile 128x128, 8 waves, wave tile 32x64 (2x4 WMMA accumulators).
// BK=64 -> 16 WMMAs per barrier pair. Weights converted f32->f16 in staging.
// OPROJ=false: out = f16 (QKV).  OPROJ=true: out = f32 with residual added.
// ---------------------------------------------------------------------------
#define BM  128
#define BN  128
#define BK  64
#define LDK 72          // 72 halfs = 144B row stride (16B multiple)

template<bool OPROJ>
__global__ __launch_bounds__(256)
void gemm_kernel(const _Float16* __restrict__ X,
                 const float* __restrict__ W,       // [N,K] row-major (torch [out,in])
                 const float* __restrict__ bias,
                 const float* __restrict__ resid,   // f32 [M,N] (OPROJ only)
                 _Float16* __restrict__ outh,       // f16 output (QKV path)
                 float* __restrict__ outf)          // f32 output (O-proj path)
{
    __shared__ _Float16 Xs[BM * LDK];
    __shared__ _Float16 Ws[BN * LDK];
    const int tid  = threadIdx.x;
    const int lane = tid & 31, wid = tid >> 5;
    const int wm = wid & 3;          // 4 wave-rows of 32
    const int wn = wid >> 2;         // 2 wave-cols of 64
    const int r  = lane & 15;
    const int cb = (lane >> 4) * 8;
    const int n0 = blockIdx.x * BN;
    const int m0 = blockIdx.y * BM;

    v8f acc[2][4] = {};

    for (int k0 = 0; k0 < DIM; k0 += BK) {
        // prefetch next K-tile (one cacheline per thread per matrix)
        if (k0 + BK < DIM) {
            __builtin_prefetch(X + (size_t)(m0 + (tid >> 1)) * DIM + k0 + BK + (tid & 1) * 32, 0, 1);
            __builtin_prefetch(W + (size_t)(n0 + (tid >> 1)) * DIM + k0 + BK + (tid & 1) * 32, 0, 1);
        }
        // stage X tile (f16, 128x64): 4 x b128 per thread
#pragma unroll
        for (int i = 0; i < 4; ++i) {
            int cid = tid + i * 256;              // 1024 chunks of 8 halfs
            int row = cid >> 3, col = (cid & 7) * 8;
            *(v8h*)&Xs[row * LDK + col] =
                *(const v8h*)(X + (size_t)(m0 + row) * DIM + k0 + col);
        }
        // stage W tile (f32 -> f16, 128x64): 8 x (b128 load + b64 ds store)
#pragma unroll
        for (int i = 0; i < 8; ++i) {
            int cid = tid + i * 256;              // 2048 chunks of 4 floats
            int row = cid >> 4, col = (cid & 15) * 4;
            v4f wv = *(const v4f*)(W + (size_t)(n0 + row) * DIM + k0 + col);
            v4h hv;
#pragma unroll
            for (int j = 0; j < 4; ++j) hv[j] = (_Float16)wv[j];
            *(v4h*)&Ws[row * LDK + col] = hv;
        }
        __syncthreads();

#pragma unroll
        for (int kc = 0; kc < 2; ++kc) {          // two 32-wide K slices
            const int kb = kc * 32 + cb;
            v16h af[2], bf[4];
#pragma unroll
            for (int mt = 0; mt < 2; ++mt) {
                const _Float16* p = &Xs[(wm * 32 + mt * 16 + r) * LDK + kb];
                af[mt] = frag_ld(p, p + 16);
            }
#pragma unroll
            for (int nt = 0; nt < 4; ++nt) {
                const _Float16* p = &Ws[(wn * 64 + nt * 16 + r) * LDK + kb];
                bf[nt] = frag_ld(p, p + 16);
            }
#pragma unroll
            for (int nt = 0; nt < 4; ++nt)
#pragma unroll
                for (int mt = 0; mt < 2; ++mt)
                    acc[mt][nt] = __builtin_amdgcn_wmma_f32_16x16x32_f16(
                        false, af[mt], false, bf[nt], (short)0, acc[mt][nt], false, false);
        }
        __syncthreads();
    }

    // epilogue: C layout => VGPR j holds rows j (lanes 0-15) / j+8 (lanes 16-31)
    const int rs = (lane >> 4) * 8;
    const int cn = lane & 15;
    float bv4[4];
#pragma unroll
    for (int nt = 0; nt < 4; ++nt) bv4[nt] = bias[n0 + wn * 64 + nt * 16 + cn];

#pragma unroll
    for (int mt = 0; mt < 2; ++mt) {
#pragma unroll
        for (int nt = 0; nt < 4; ++nt) {
            size_t idx = (size_t)(m0 + wm * 32 + mt * 16 + rs) * DIM
                       + (size_t)(n0 + wn * 64 + nt * 16 + cn);
#pragma unroll
            for (int j = 0; j < 8; ++j, idx += DIM) {
                float v = acc[mt][nt][j] + bv4[nt];
                if constexpr (OPROJ) {
                    outf[idx] = v + resid[idx];
                } else {
                    outh[idx] = (_Float16)v;
                }
            }
        }
    }
}

// ---------------------------------------------------------------------------
// Flash attention: one block per (batch, head, 64-query tile), 4 waves,
// wave owns 16 query rows. Online softmax over Nc in chunks of 64.
// K chunk staged by the Tensor Data Mover (with HW padding into LDC=72
// layout); V staged manually with transpose; row sums of P via P @ ones.
// ---------------------------------------------------------------------------
#define CT  64
#define LDC 72          // 72 halfs = 144B stride (16B multiple)

__global__ __launch_bounds__(128)
void attn_kernel(const _Float16* __restrict__ Qh,
                 const _Float16* __restrict__ Kh,
                 const _Float16* __restrict__ Vh,
                 _Float16* __restrict__ Ah)
{
    __shared__ _Float16 Ks[CT * LDC];   // K chunk, row-major [c][d] (TDM-filled)
    __shared__ _Float16 Vt[HD * LDC];   // V chunk transposed [d][c]
    __shared__ _Float16 Ps[64 * LDC];   // probabilities, [q][c]

    const int tid  = threadIdx.x;
    const int lane = tid & 31, w = tid >> 5;
    const int r  = lane & 15;
    const int cb = (lane >> 4) * 8;
    const int rs = cb;
    const int cn = r;
    const int q0 = blockIdx.x * 64;
    const int h  = blockIdx.y;
    const int b  = blockIdx.z;

    // Q fragments for this wave's 16 rows (K-dim = 64 -> 2 frags), kept in regs
    const size_t qrow = (size_t)(b * NQ + q0 + w * 16 + r) * DIM + h * HD;
    v16h qf[2];
#pragma unroll
    for (int kc = 0; kc < 2; ++kc)
        qf[kc] = frag_ld(Qh + qrow + kc * 32 + cb, Qh + qrow + kc * 32 + cb + 16);

    // all-ones B fragment: D = P @ ones puts row-sum(P) in every column
    v16h onesf;
#pragma unroll
    for (int i = 0; i < 16; ++i) onesf[i] = (_Float16)1.0f;

    // Tensor DMA descriptor (constant part): 2D tile 64x64 halfs, tensor row
    // stride 1024 halfs, LDS padding 4 DWORDs every 32 DWORDs (-> LDC=72).
    const unsigned ks_lds = (unsigned)(unsigned long long)(&Ks[0]);
    v8i g1;
    g1[0] = (1 << 16) | (1 << 20) | (4 << 22) | (3 << 25); // data_size=2B, pad_en, intv=32dw, amt=4dw
    g1[1] = (int)(64u << 16);     // tensor_dim0 = 64 (halfs per tile row)
    g1[2] = (int)(64u << 16);     // tensor_dim1 = 64 rows
    g1[3] = (int)(64u << 16);     // tile_dim0 = 64
    g1[4] = 64;                   // tile_dim1 = 64
    g1[5] = 1024;                 // tensor_dim0_stride = DIM halfs
    g1[6] = 0; g1[7] = 0;
    const v4i zg4 = {};
    const v8i zg8 = {};

    float mrun[8], lrun[8];
#pragma unroll
    for (int j = 0; j < 8; ++j) { mrun[j] = -3.0e38f; lrun[j] = 0.f; }
    v8f oacc[4] = {};
    const float scale = 0.125f;   // 1/sqrt(64)

    for (int c0 = 0; c0 < NC; c0 += CT) {
        // K chunk: Tensor Data Mover, issued by wave 0 only (EXEC-independent)
        if (w == 0) {
            unsigned long long ga =
                (unsigned long long)(Kh + (size_t)(b * NC + c0) * DIM + h * HD);
            v4u g0;
            g0[0] = 1u;                                  // count=1 (valid user D#)
            g0[1] = ks_lds;                              // LDS dest byte address
            g0[2] = (unsigned)(ga & 0xffffffffu);        // global_addr[31:0]
            g0[3] = (unsigned)(ga >> 32) | (2u << 30);   // global_addr[56:32] | type=2
            __builtin_amdgcn_tensor_load_to_lds(g0, g1, zg4, zg4, zg8, 0);
        }
        // prefetch next chunk's V rows; stage V chunk transposed into LDS
        if (c0 + CT < NC) {
            size_t nb = (size_t)(b * NC + c0 + CT + (tid >> 1)) * DIM + h * HD + (tid & 1) * 32;
            __builtin_prefetch(Vh + nb, 0, 1);
        }
#pragma unroll
        for (int i = 0; i < 4; ++i) {
            int cid = tid + i * 128;
            int row = cid >> 3, col = (cid & 7) * 8;
            v8h vv = *(const v8h*)(Vh + (size_t)(b * NC + c0 + row) * DIM + h * HD + col);
#pragma unroll
            for (int j = 0; j < 8; ++j) Vt[(col + j) * LDC + row] = vv[j];
        }
        if (w == 0) __builtin_amdgcn_s_wait_tensorcnt(0);
        __syncthreads();

        // S = Q @ K^T : preload 4 B-fragments per K-slice, then WMMA burst
        v8f s[4] = {};
#pragma unroll
        for (int kc = 0; kc < 2; ++kc) {
            v16h bf[4];
#pragma unroll
            for (int t = 0; t < 4; ++t) {
                const _Float16* p = &Ks[(t * 16 + r) * LDC + kc * 32 + cb];
                bf[t] = frag_ld(p, p + 16);
            }
#pragma unroll
            for (int t = 0; t < 4; ++t)
                s[t] = __builtin_amdgcn_wmma_f32_16x16x32_f16(
                    false, qf[kc], false, bf[t], (short)0, s[t], false, false);
        }
#pragma unroll
        for (int t = 0; t < 4; ++t)
#pragma unroll
            for (int j = 0; j < 8; ++j) s[t][j] *= scale;

        // online softmax: row max via xor-butterfly over the 16-lane half
        float corr[8], mnew[8];
#pragma unroll
        for (int j = 0; j < 8; ++j) {
            float v = fmaxf(fmaxf(s[0][j], s[1][j]), fmaxf(s[2][j], s[3][j]));
#pragma unroll
            for (int off = 1; off < 16; off <<= 1) v = fmaxf(v, __shfl_xor(v, off, 32));
            mnew[j] = fmaxf(mrun[j], v);
            corr[j] = __expf(mrun[j] - mnew[j]);
            mrun[j] = mnew[j];
        }
        // exponentiate; rescale O; spill P (C-layout) to LDS as row-major f16
#pragma unroll
        for (int t = 0; t < 4; ++t)
#pragma unroll
            for (int j = 0; j < 8; ++j) {
                float p = __expf(s[t][j] - mnew[j]);
                s[t][j] = p;
                oacc[t][j] *= corr[j];
                Ps[(w * 16 + j + rs) * LDC + t * 16 + cn] = (_Float16)p;
            }

        // reload P as A-fragments; row sums via P @ ones; O += P @ V
        v16h pf[2];
#pragma unroll
        for (int kc = 0; kc < 2; ++kc) {
            const _Float16* pp = &Ps[(w * 16 + r) * LDC + kc * 32 + cb];
            pf[kc] = frag_ld(pp, pp + 16);
        }
        v8f rsum = __builtin_amdgcn_wmma_f32_16x16x32_f16(
            false, pf[0], false, onesf, (short)0, v8f{}, false, false);
        rsum = __builtin_amdgcn_wmma_f32_16x16x32_f16(
            false, pf[1], false, onesf, (short)0, rsum, false, false);
#pragma unroll
        for (int j = 0; j < 8; ++j) lrun[j] = lrun[j] * corr[j] + rsum[j];

#pragma unroll
        for (int kc = 0; kc < 2; ++kc) {
            v16h vf[4];
#pragma unroll
            for (int t = 0; t < 4; ++t) {
                const _Float16* vp = &Vt[(t * 16 + r) * LDC + kc * 32 + cb];
                vf[t] = frag_ld(vp, vp + 16);
            }
#pragma unroll
            for (int t = 0; t < 4; ++t)
                oacc[t] = __builtin_amdgcn_wmma_f32_16x16x32_f16(
                    false, pf[kc], false, vf[t], (short)0, oacc[t], false, false);
        }
        __syncthreads();
    }

    // normalize and write attention output (f16, head-concatenated layout)
#pragma unroll
    for (int j = 0; j < 8; ++j) lrun[j] = 1.0f / lrun[j];
#pragma unroll
    for (int t = 0; t < 4; ++t)
#pragma unroll
        for (int j = 0; j < 8; ++j) {
            size_t gm = (size_t)(b * NQ + q0 + w * 16 + j + rs) * DIM + h * HD + t * 16 + cn;
            Ah[gm] = (_Float16)(oacc[t][j] * lrun[j]);
        }
}

// ---------------------------------------------------------------------------
extern "C" void kernel_launch(void* const* d_in, const int* in_sizes, int n_in,
                              void* d_out, int out_size, void* d_ws, size_t ws_size,
                              hipStream_t stream)
{
    (void)in_sizes; (void)n_in; (void)out_size; (void)ws_size;
    const float* query   = (const float*)d_in[0];
    const float* context = (const float*)d_in[1];
    const float* wq = (const float*)d_in[2];
    const float* bq = (const float*)d_in[3];
    const float* wk = (const float*)d_in[4];
    const float* bk = (const float*)d_in[5];
    const float* wv = (const float*)d_in[6];
    const float* bv = (const float*)d_in[7];
    const float* wo = (const float*)d_in[8];
    const float* bo = (const float*)d_in[9];
    const float* g_q  = (const float*)d_in[10];
    const float* b_q  = (const float*)d_in[11];
    const float* g_kv = (const float*)d_in[12];
    const float* b_kv = (const float*)d_in[13];

    char* ws = (char*)d_ws;                                    // 68 MB used
    float*    qn32 = (float*)(ws);                             //  8 MB [2048,1024] f32
    _Float16* qnh  = (_Float16*)(ws + (size_t)8  * (1 << 20)); //  4 MB
    _Float16* cnh  = (_Float16*)(ws + (size_t)12 * (1 << 20)); // 16 MB
    _Float16* Qh   = (_Float16*)(ws + (size_t)28 * (1 << 20)); //  4 MB
    _Float16* Kh   = (_Float16*)(ws + (size_t)32 * (1 << 20)); // 16 MB
    _Float16* Vh   = (_Float16*)(ws + (size_t)48 * (1 << 20)); // 16 MB
    _Float16* Ah   = (_Float16*)(ws + (size_t)64 * (1 << 20)); //  4 MB

    // LayerNorm + cast
    ln_kernel<<<2 * NQ, 256, 0, stream>>>(query,   g_q,  b_q,  qn32,    qnh);
    ln_kernel<<<2 * NC, 256, 0, stream>>>(context, g_kv, b_kv, nullptr, cnh);

    // Q/K/V projections (f16 WMMA, f32 accumulate)
    gemm_kernel<false><<<dim3(DIM / BN, (2 * NQ) / BM), 256, 0, stream>>>(
        qnh, wq, bq, nullptr, Qh, nullptr);
    gemm_kernel<false><<<dim3(DIM / BN, (2 * NC) / BM), 256, 0, stream>>>(
        cnh, wk, bk, nullptr, Kh, nullptr);
    gemm_kernel<false><<<dim3(DIM / BN, (2 * NC) / BM), 256, 0, stream>>>(
        cnh, wv, bv, nullptr, Vh, nullptr);

    // Flash attention (TDM-staged K)
    attn_kernel<<<dim3(NQ / 64, NHEADS, 2), 128, 0, stream>>>(Qh, Kh, Vh, Ah);

    // Output projection + bias + residual (qn) -> f32 d_out
    gemm_kernel<true><<<dim3(DIM / BN, (2 * NQ) / BM), 256, 0, stream>>>(
        Ah, wo, bo, qn32, nullptr, (float*)d_out);
}
